// MultiHeadAttention_20796231647734
// MI455X (gfx1250) — compile-verified, matrix operand order checked by reading, not run on hardware
//
#include <hip/hip_runtime.h>

typedef __attribute__((ext_vector_type(16))) __bf16 v16bf;
typedef __attribute__((ext_vector_type(8)))  __bf16 v8bf;
typedef __attribute__((ext_vector_type(8)))  float  v8f;

#define NB   4
#define NT   2048
#define NC   1024
#define NH   16
#define NHD  64
#define N3C  3072

__device__ __forceinline__ v8f wmma_bf16(v16bf a, v16bf b, v8f c) {
  // D = A(16x32 bf16) * B(32x16 bf16) + C(16x16 f32)
  return __builtin_amdgcn_wmma_f32_16x16x32_bf16(false, a, false, b, (short)0, c,
                                                 false, false);
}

// Load a 16(lane-dim) x 32(K) bf16 fragment.
// p points to (lane-row 0, k=0); lane rows strided by ld elements, K contiguous.
// CDNA5 16-bit A/B per-lane layout: lane l -> row l&15, half = l>>4;
//   elems 0..7  hold k = half*8 + i
//   elems 8..15 hold k = 16 + half*8 + (i-8)
// => two contiguous 16-byte loads per lane (global_load_b128 x2).
__device__ __forceinline__ v16bf ldfrag(const __bf16* p, int ld) {
  int lane = threadIdx.x & 31;
  int half = lane >> 4;
  const v8bf* pv = (const v8bf*)(p + (size_t)(lane & 15) * (size_t)ld);
  v8bf lo = pv[half];
  v8bf hi = pv[2 + half];
  v16bf f;
#pragma unroll
  for (int i = 0; i < 8; ++i) { f[i] = lo[i]; f[i + 8] = hi[i]; }
  return f;
}

__device__ __forceinline__ float rowmax16(float v) {
  v = fmaxf(v, __shfl_xor(v, 1, 32));
  v = fmaxf(v, __shfl_xor(v, 2, 32));
  v = fmaxf(v, __shfl_xor(v, 4, 32));
  v = fmaxf(v, __shfl_xor(v, 8, 32));
  return v;
}
__device__ __forceinline__ float rowsum16(float v) {
  v += __shfl_xor(v, 1, 32);
  v += __shfl_xor(v, 2, 32);
  v += __shfl_xor(v, 4, 32);
  v += __shfl_xor(v, 8, 32);
  return v;
}

__global__ void cvt_f32_bf16(const float* __restrict__ src,
                             __bf16* __restrict__ dst, int n) {
  int i = blockIdx.x * blockDim.x + threadIdx.x;
  if (i < n) dst[i] = (__bf16)src[i];
}

// qkv = x @ w_qkv^T : M=8192, N=3072, K=1024. One wave per 32x32 output tile.
// Output scattered into q [B,H,T,hd], k [B,H,T,hd], vT [B,H,hd,T] (bf16).
__global__ __launch_bounds__(128) void qkv_gemm(const __bf16* __restrict__ xb,
                                                const __bf16* __restrict__ wb,
                                                __bf16* __restrict__ qo,
                                                __bf16* __restrict__ ko,
                                                __bf16* __restrict__ vto) {
  const int wave = threadIdx.x >> 5;
  const int lane = threadIdx.x & 31;
  const int half = lane >> 4, nl = lane & 15;
  const int m0 = blockIdx.y * 32;                 // 0..8160
  const int n0 = (blockIdx.x * 4 + wave) * 32;    // 0..3040

  v8f acc[2][2] = {};
  for (int k0 = 0; k0 < NC; k0 += 32) {
    v16bf a0 = ldfrag(xb + (size_t)m0 * NC + k0, NC);
    v16bf a1 = ldfrag(xb + (size_t)(m0 + 16) * NC + k0, NC);
    v16bf b0 = ldfrag(wb + (size_t)n0 * NC + k0, NC);
    v16bf b1 = ldfrag(wb + (size_t)(n0 + 16) * NC + k0, NC);
    acc[0][0] = wmma_bf16(a0, b0, acc[0][0]);
    acc[0][1] = wmma_bf16(a0, b1, acc[0][1]);
    acc[1][0] = wmma_bf16(a1, b0, acc[1][0]);
    acc[1][1] = wmma_bf16(a1, b1, acc[1][1]);
  }
#pragma unroll
  for (int i = 0; i < 2; ++i) {
#pragma unroll
    for (int j = 0; j < 2; ++j) {
#pragma unroll
      for (int r = 0; r < 8; ++r) {
        int m = m0 + i * 16 + r + half * 8;       // row in [0,8192)
        int n = n0 + j * 16 + nl;                 // col in [0,3072)
        int b = m >> 11, t = m & (NT - 1);
        int which = n >> 10;                      // 0=q 1=k 2=v
        int c = n & (NC - 1);
        int h = c >> 6, d = c & (NHD - 1);
        __bf16 bv = (__bf16)acc[i][j][r];
        size_t bh = (size_t)(b * NH + h);
        if (which == 0)      qo[(bh * NT + t) * NHD + d] = bv;
        else if (which == 1) ko[(bh * NT + t) * NHD + d] = bv;
        else                 vto[(bh * NHD + d) * NT + t] = bv;
      }
    }
  }
}

// Flash attention: one wave handles one (b,h, 16-query tile). 32 keys/iter.
__global__ __launch_bounds__(128) void flash_attn(const __bf16* __restrict__ q,
                                                  const __bf16* __restrict__ k,
                                                  const __bf16* __restrict__ vt,
                                                  __bf16* __restrict__ ao) {
  __shared__ __align__(16) __bf16 plds[4][16 * 32];
  const int wave = threadIdx.x >> 5;
  const int lane = threadIdx.x & 31;
  const int half = lane >> 4, nl = lane & 15;
  const int task = blockIdx.x * 4 + wave;   // 0..8191
  const int qt = task & 127;                // query tile in [0, T/16)
  const int bh = task >> 7;                 // 0..63
  const int qbase = qt * 16;

  const __bf16* qp = q  + (size_t)bh * NT * NHD;
  const __bf16* kp = k  + (size_t)bh * NT * NHD;
  const __bf16* vp = vt + (size_t)bh * NHD * NT;

  v16bf qa0 = ldfrag(qp + (size_t)qbase * NHD +  0, NHD);
  v16bf qa1 = ldfrag(qp + (size_t)qbase * NHD + 32, NHD);

  v8f o[4] = {};
  float mrun[8], lrun[8];
#pragma unroll
  for (int r = 0; r < 8; ++r) { mrun[r] = -1e30f; lrun[r] = 0.0f; }

  const float scale = 0.125f;               // 1/sqrt(64)
  const int nkt = qbase / 32 + 1;           // causal: tiles up to the diagonal
  for (int kt = 0; kt < nkt; ++kt) {
    const int kbase = kt * 32;
    // S = Q K^T  (two 16x16 key sub-tiles, hd=64 -> 2 WMMA K-steps each)
    v8f s0 = {}, s1 = {};
    s0 = wmma_bf16(qa0, ldfrag(kp + (size_t)kbase * NHD +  0, NHD), s0);
    s0 = wmma_bf16(qa1, ldfrag(kp + (size_t)kbase * NHD + 32, NHD), s0);
    s1 = wmma_bf16(qa0, ldfrag(kp + (size_t)(kbase + 16) * NHD +  0, NHD), s1);
    s1 = wmma_bf16(qa1, ldfrag(kp + (size_t)(kbase + 16) * NHD + 32, NHD), s1);

    // scale + causal mask (D-layout: row = r + half*8, col = nl)
#pragma unroll
    for (int r = 0; r < 8; ++r) {
      int qrow = qbase + r + half * 8;
      float a = s0[r] * scale, b = s1[r] * scale;
      if (kbase + nl > qrow)      a = -1e30f;
      if (kbase + 16 + nl > qrow) b = -1e30f;
      s0[r] = a; s1[r] = b;
    }
    // online softmax (row reductions across each 16-lane half)
#pragma unroll
    for (int r = 0; r < 8; ++r) {
      float tmax = rowmax16(fmaxf(s0[r], s1[r]));
      float mnew = fmaxf(mrun[r], tmax);
      float alpha = __expf(mrun[r] - mnew);
      float p0 = __expf(s0[r] - mnew);
      float p1 = __expf(s1[r] - mnew);
      lrun[r] = lrun[r] * alpha + rowsum16(p0 + p1);
      mrun[r] = mnew;
      s0[r] = p0; s1[r] = p1;
      o[0][r] *= alpha; o[1][r] *= alpha; o[2][r] *= alpha; o[3][r] *= alpha;
    }
    // re-layout P (D-layout f32) -> A-fragment bf16 through per-wave LDS tile
    __bf16* pw = &plds[wave][0];
#pragma unroll
    for (int r = 0; r < 8; ++r) {
      int row = r + half * 8;
      pw[row * 32 + nl]      = (__bf16)s0[r];
      pw[row * 32 + 16 + nl] = (__bf16)s1[r];
    }
    v16bf pa;
    {
      const v8bf* pv = (const v8bf*)(pw + nl * 32);
      v8bf lo = pv[half];
      v8bf hi = pv[2 + half];
#pragma unroll
      for (int i = 0; i < 8; ++i) { pa[i] = lo[i]; pa[i + 8] = hi[i]; }
    }
    // O += P V  (V^T layout: rows = hd, contiguous along keys)
#pragma unroll
    for (int d = 0; d < 4; ++d) {
      v16bf vb = ldfrag(vp + (size_t)(d * 16) * NT + kbase, NT);
      o[d] = wmma_bf16(pa, vb, o[d]);
    }
  }

  // normalize + write attention output into [B,T,C] bf16
  const int b = bh >> 4, h = bh & 15;
#pragma unroll
  for (int r = 0; r < 8; ++r) {
    float inv = 1.0f / lrun[r];
    int t = qbase + r + half * 8;
    size_t rowoff = ((size_t)(b * NT + t)) * NC + h * NHD;
#pragma unroll
    for (int d = 0; d < 4; ++d)
      ao[rowoff + d * 16 + nl] = (__bf16)(o[d][r] * inv);
  }
}

// out = ao @ w_out^T : M=8192, N=1024, K=1024, fp32 output in [B,T,C].
__global__ __launch_bounds__(128) void out_gemm(const __bf16* __restrict__ ao,
                                                const __bf16* __restrict__ wo,
                                                float* __restrict__ out) {
  const int wave = threadIdx.x >> 5;
  const int lane = threadIdx.x & 31;
  const int half = lane >> 4, nl = lane & 15;
  const int m0 = blockIdx.y * 32;
  const int n0 = (blockIdx.x * 4 + wave) * 32;

  v8f acc[2][2] = {};
  for (int k0 = 0; k0 < NC; k0 += 32) {
    v16bf a0 = ldfrag(ao + (size_t)m0 * NC + k0, NC);
    v16bf a1 = ldfrag(ao + (size_t)(m0 + 16) * NC + k0, NC);
    v16bf b0 = ldfrag(wo + (size_t)n0 * NC + k0, NC);
    v16bf b1 = ldfrag(wo + (size_t)(n0 + 16) * NC + k0, NC);
    acc[0][0] = wmma_bf16(a0, b0, acc[0][0]);
    acc[0][1] = wmma_bf16(a0, b1, acc[0][1]);
    acc[1][0] = wmma_bf16(a1, b0, acc[1][0]);
    acc[1][1] = wmma_bf16(a1, b1, acc[1][1]);
  }
#pragma unroll
  for (int i = 0; i < 2; ++i)
#pragma unroll
    for (int j = 0; j < 2; ++j)
#pragma unroll
      for (int r = 0; r < 8; ++r) {
        int m = m0 + i * 16 + r + half * 8;
        int n = n0 + j * 16 + nl;
        out[(size_t)m * NC + n] = acc[i][j][r];
      }
}

extern "C" void kernel_launch(void* const* d_in, const int* in_sizes, int n_in,
                              void* d_out, int out_size, void* d_ws, size_t ws_size,
                              hipStream_t stream) {
  const float* x     = (const float*)d_in[0];   // [4,2048,1024]
  const float* w_qkv = (const float*)d_in[1];   // [3072,1024]
  const float* w_out = (const float*)d_in[2];   // [1024,1024]
  float* out = (float*)d_out;                   // [4,2048,1024]

  char* ws = (char*)d_ws;
  const size_t xN = (size_t)NB * NT * NC;       // 8,388,608
  const size_t wqN = (size_t)N3C * NC;          // 3,145,728
  const size_t woN = (size_t)NC * NC;           // 1,048,576
  const size_t qkvN = (size_t)NB * NH * NT * NHD; // per-tensor 8,388,608

  __bf16* xb    = (__bf16*)ws;                 ws += xN * 2;     // also reused as ao
  __bf16* wqkvb = (__bf16*)ws;                 ws += wqN * 2;
  __bf16* woutb = (__bf16*)ws;                 ws += woN * 2;
  __bf16* qb    = (__bf16*)ws;                 ws += qkvN * 2;
  __bf16* kb    = (__bf16*)ws;                 ws += qkvN * 2;
  __bf16* vtb   = (__bf16*)ws;                 ws += qkvN * 2;
  __bf16* aob   = xb;  // x no longer needed after qkv_gemm

  cvt_f32_bf16<<<(int)((xN + 255) / 256), 256, 0, stream>>>(x, xb, (int)xN);
  cvt_f32_bf16<<<(int)((wqN + 255) / 256), 256, 0, stream>>>(w_qkv, wqkvb, (int)wqN);
  cvt_f32_bf16<<<(int)((woN + 255) / 256), 256, 0, stream>>>(w_out, woutb, (int)woN);

  // M=8192 -> 256 tiles of 32; N=3072 -> 96 tiles of 32, 4 waves/block
  qkv_gemm<<<dim3(24, 256, 1), 128, 0, stream>>>(xb, wqkvb, qb, kb, vtb);

  // 64 (b,h) * 128 query tiles = 8192 waves, 4 waves/block
  flash_attn<<<2048, 128, 0, stream>>>(qb, kb, vtb, aob);

  // M=8192 -> 256 tiles; N=1024 -> 32 tiles, 4 waves/block
  out_gemm<<<dim3(8, 256, 1), 128, 0, stream>>>(aob, woutb, out);
}
